// LSTM_4157528343129
// MI455X (gfx1250) — compile-verified, matrix operand order checked by reading, not run on hardware
//
#include <hip/hip_runtime.h>
#include <cstdint>
#include <cstddef>

// Problem constants (from reference): B=64, L=512, E=1024, H=1024
#define B_ 64
#define L_ 512
#define E_ 1024
#define H_ 1024
#define G_ 4096   // 4*H (gates f,i,o,c concatenated)

// ---------------------------------------------------------------------------
// CDNA5 WMMA types: 16x16x32 bf16 -> f32
// ---------------------------------------------------------------------------
typedef __attribute__((ext_vector_type(16))) __bf16 v16bf;
typedef __attribute__((ext_vector_type(8)))  float  v8f;

union BFrag { v16bf v; uint4 q[2]; };
union FAcc  { v8f  v; float f[8]; };

// ---------------------------------------------------------------------------
// Async global->LDS DMA (gfx1250). Builtin signature (from clang diagnostic):
//   void __builtin_amdgcn_global_load_async_to_lds_b128(
//       AS(1) v4i* src, AS(3) v4i* dst, imm int offset, imm int cpol)
// ---------------------------------------------------------------------------
typedef int v4i_ __attribute__((vector_size(4 * sizeof(int))));

#if defined(__HIP_DEVICE_COMPILE__) && \
    __has_builtin(__builtin_amdgcn_global_load_async_to_lds_b128) && \
    __has_builtin(__builtin_amdgcn_s_wait_asynccnt)
#define USE_ASYNC_LDS 1
#define ASYNC_CP16(gp, lp)                                                  \
    __builtin_amdgcn_global_load_async_to_lds_b128(                         \
        (__attribute__((address_space(1))) v4i_*)(v4i_*)(gp),               \
        (__attribute__((address_space(3))) v4i_*)(v4i_*)(lp), 0, 0)
#define ASYNC_WAIT() __builtin_amdgcn_s_wait_asynccnt(0)
#else
#define USE_ASYNC_LDS 0
#endif

// ---------------------------------------------------------------------------
// Pack kernels: f32 -> bf16, gate-concatenated weights [4H, K]
// ---------------------------------------------------------------------------
__global__ void pack_x_kernel(const float* __restrict__ x,
                              __bf16* __restrict__ xb, int n) {
    for (int i = blockIdx.x * blockDim.x + threadIdx.x; i < n;
         i += gridDim.x * blockDim.x)
        xb[i] = (__bf16)x[i];
}

__global__ void pack_wb_kernel(const float* __restrict__ Wfx, const float* __restrict__ Wix,
                               const float* __restrict__ Wox, const float* __restrict__ Wcx,
                               const float* __restrict__ Wfm, const float* __restrict__ Wim,
                               const float* __restrict__ Wom, const float* __restrict__ Wcm,
                               const float* __restrict__ bfx, const float* __restrict__ bix,
                               const float* __restrict__ box, const float* __restrict__ bcx,
                               const float* __restrict__ bfm, const float* __restrict__ bim,
                               const float* __restrict__ bom, const float* __restrict__ bcm,
                               __bf16* __restrict__ WXb, __bf16* __restrict__ WMb,
                               float* __restrict__ bx4, float* __restrict__ bm4) {
    const float* WX[4] = {Wfx, Wix, Wox, Wcx};
    const float* WM[4] = {Wfm, Wim, Wom, Wcm};
    const float* BX[4] = {bfx, bix, box, bcx};
    const float* BM[4] = {bfm, bim, bom, bcm};
    const int n = 4 * H_ * E_;
    for (int o = blockIdx.x * blockDim.x + threadIdx.x; o < n;
         o += gridDim.x * blockDim.x) {
        int g = o / (H_ * E_);
        int rem = o - g * (H_ * E_);
        WXb[o] = (__bf16)WX[g][rem];
        WMb[o] = (__bf16)WM[g][rem];
        if (o < G_) {
            int gg = o / H_;
            int j  = o - gg * H_;
            bx4[o] = BX[gg][j];
            bm4[o] = BM[gg][j];
        }
    }
}

__global__ void init_state_kernel(float* __restrict__ hF, float* __restrict__ cF,
                                  __bf16* __restrict__ hB0, __bf16* __restrict__ hB1,
                                  int n) {
    for (int i = blockIdx.x * blockDim.x + threadIdx.x; i < n;
         i += gridDim.x * blockDim.x) {
        hF[i] = 0.0f;
        cF[i] = 0.0f;
        hB0[i] = (__bf16)0.0f;
        hB1[i] = (__bf16)0.0f;
    }
}

// ---------------------------------------------------------------------------
// Phase 1: xproj = X @ Wx^T + bx  (M=B*L=32768, N=4H=4096, K=E=1024)
// Output stored time-major: XP[(l*B + b)*4096 + n].
// WG: 256 threads = 8 waves (2 along M x 4 along N), 128x128 tile, K staged
// 32 at a time. Async DMA double-buffers LDS against the WMMA stream.
// ---------------------------------------------------------------------------
__global__ __launch_bounds__(256)
void gemm_xproj_kernel(const __bf16* __restrict__ X,   // [B*L, E]
                       const __bf16* __restrict__ W,   // [4H, E]
                       const float*  __restrict__ bx4, // [4H]
                       float* __restrict__ XP)         // [L, B, 4H]
{
    __shared__ __bf16 As[2][128][32];
    __shared__ __bf16 Bs[2][128][32];

    const int tid   = threadIdx.x;
    const int lane  = tid & 31;
    const int wid   = tid >> 5;
    const int wm    = wid & 1;   // 0..1 (M)
    const int wn    = wid >> 1;  // 0..3 (N)
    const int mtile = blockIdx.x * 128;
    const int ntile = blockIdx.y * 128;

    const int row16 = lane & 15;
    const int kh    = lane >> 4;   // K-half select per ISA bf16 A/B layout

    FAcc acc[4][2] = {};

    // staging map: 512 x 16B chunks per tile, 2 per thread for A and B
    const int c0 = tid, c1 = tid + 256;
    const int r0 = c0 >> 2, s0 = c0 & 3;
    const int r1 = c1 >> 2, s1 = c1 & 3;
    const __bf16* gA0 = X + (size_t)(mtile + r0) * E_ + s0 * 8;
    const __bf16* gA1 = X + (size_t)(mtile + r1) * E_ + s1 * 8;
    const __bf16* gB0 = W + (size_t)(ntile + r0) * E_ + s0 * 8;
    const __bf16* gB1 = W + (size_t)(ntile + r1) * E_ + s1 * 8;

#if USE_ASYNC_LDS
    // prologue: DMA K-tile 0 into buffer 0
    ASYNC_CP16(gA0, &As[0][r0][s0 * 8]);
    ASYNC_CP16(gA1, &As[0][r1][s1 * 8]);
    ASYNC_CP16(gB0, &Bs[0][r0][s0 * 8]);
    ASYNC_CP16(gB1, &Bs[0][r1][s1 * 8]);
    ASYNC_WAIT();
    __syncthreads();
#endif

    for (int k = 0; k < E_ / 32; ++k) {
        const int kk  = k * 32;
        const int buf = k & 1;

#if USE_ASYNC_LDS
        // DMA next K-tile into the other buffer while this one computes
        if (kk + 32 < E_) {
            ASYNC_CP16(gA0 + kk + 32, &As[buf ^ 1][r0][s0 * 8]);
            ASYNC_CP16(gA1 + kk + 32, &As[buf ^ 1][r1][s1 * 8]);
            ASYNC_CP16(gB0 + kk + 32, &Bs[buf ^ 1][r0][s0 * 8]);
            ASYNC_CP16(gB1 + kk + 32, &Bs[buf ^ 1][r1][s1 * 8]);
        }
#else
        uint4 a0 = *(const uint4*)(gA0 + kk);
        uint4 a1 = *(const uint4*)(gA1 + kk);
        uint4 b0 = *(const uint4*)(gB0 + kk);
        uint4 b1 = *(const uint4*)(gB1 + kk);
        if (kk + 32 < E_) {
            __builtin_prefetch(gA0 + kk + 32, 0, 1);
            __builtin_prefetch(gB0 + kk + 32, 0, 1);
        }
        __syncthreads();
        *(uint4*)&As[0][r0][s0 * 8] = a0;
        *(uint4*)&As[0][r1][s1 * 8] = a1;
        *(uint4*)&Bs[0][r0][s0 * 8] = b0;
        *(uint4*)&Bs[0][r1][s1 * 8] = b1;
        __syncthreads();
#endif

        const int sb = USE_ASYNC_LDS ? buf : 0;
        BFrag fa[4], fb[2];
        #pragma unroll
        for (int mf = 0; mf < 4; ++mf) {
            const __bf16* ap = &As[sb][wm * 64 + mf * 16 + row16][kh * 8];
            fa[mf].q[0] = *(const uint4*)ap;        // K = kh*8 + 0..7
            fa[mf].q[1] = *(const uint4*)(ap + 16); // K = kh*8 + 16..23
        }
        #pragma unroll
        for (int nf = 0; nf < 2; ++nf) {
            const __bf16* bp = &Bs[sb][wn * 32 + nf * 16 + row16][kh * 8];
            fb[nf].q[0] = *(const uint4*)bp;
            fb[nf].q[1] = *(const uint4*)(bp + 16);
        }
        #pragma unroll
        for (int mf = 0; mf < 4; ++mf)
            #pragma unroll
            for (int nf = 0; nf < 2; ++nf)
                acc[mf][nf].v = __builtin_amdgcn_wmma_f32_16x16x32_bf16(
                    false, fa[mf].v, false, fb[nf].v,
                    (short)0, acc[mf][nf].v, false, false);

#if USE_ASYNC_LDS
        ASYNC_WAIT();      // next tile fully in LDS
        __syncthreads();   // ... and visible to all waves; this tile consumed
#endif
    }

    // C/D layout: lane = N + 16*(M>=8); vgpr r -> M = (lane>>4)*8 + r
    const int nloc = lane & 15;
    const int mhi  = lane >> 4;
    #pragma unroll
    for (int mf = 0; mf < 4; ++mf) {
        #pragma unroll
        for (int nf = 0; nf < 2; ++nf) {
            const int col  = ntile + wn * 32 + nf * 16 + nloc;
            const float bias = bx4[col];
            #pragma unroll
            for (int r = 0; r < 8; ++r) {
                const int rowg = mtile + wm * 64 + mf * 16 + mhi * 8 + r;
                const int b = rowg >> 9;        // rowg / L
                const int l = rowg & (L_ - 1);  // rowg % L
                XP[((size_t)(l * B_ + b) << 12) + col] = acc[mf][nf].f[r] + bias;
            }
        }
    }
}

// ---------------------------------------------------------------------------
// Phase 2: one recurrent timestep.
// Grid = 64 WGs x 16 hidden units each; block = 4 waves, wave w = gate w.
// h (full [64,1024] bf16) is K-streamed through LDS via async DMA; Wm rows
// feed B fragments straight from L2. h ping-pongs across two global buffers.
// ---------------------------------------------------------------------------
__global__ __launch_bounds__(128)
void lstm_step_kernel(const float*  __restrict__ XP,    // [L, B, 4H]
                      const __bf16* __restrict__ WM,    // [4H, H]
                      const float*  __restrict__ bm4,   // [4H]
                      const int*    __restrict__ maskBL,// [B, L]
                      const __bf16* __restrict__ hBin,  // [B, H]
                      __bf16*       __restrict__ hBout, // [B, H]
                      float*        __restrict__ hF,    // [B, H]
                      float*        __restrict__ cF,    // [B, H]
                      float*        __restrict__ out,   // [B, L, H]
                      int t)
{
    __shared__ __bf16 hs[2][64][32];
    __shared__ float  gbuf[4][64][16];

    const int tid   = threadIdx.x;
    const int lane  = tid & 31;
    const int wid   = tid >> 5;           // gate index: 0=f 1=i 2=o 3=c
    const int jbase = blockIdx.x * 16;

    const int row16 = lane & 15;
    const int kh    = lane >> 4;

    FAcc acc[4] = {};   // 4 M-frags covering B=64 rows

    const __bf16* wrow = WM + (size_t)(wid * H_ + jbase + row16) * H_;

    // staging map: 256 x 16B chunks per tile, 2 per thread
    const int c0 = tid, c1 = tid + 128;
    const int r0 = c0 >> 2, s0 = c0 & 3;
    const int r1 = c1 >> 2, s1 = c1 & 3;
    const __bf16* gH0 = hBin + (size_t)r0 * H_ + s0 * 8;
    const __bf16* gH1 = hBin + (size_t)r1 * H_ + s1 * 8;

#if USE_ASYNC_LDS
    ASYNC_CP16(gH0, &hs[0][r0][s0 * 8]);
    ASYNC_CP16(gH1, &hs[0][r1][s1 * 8]);
    ASYNC_WAIT();
    __syncthreads();
#endif

    for (int k = 0; k < H_ / 32; ++k) {
        const int kk  = k * 32;
        const int buf = k & 1;

#if USE_ASYNC_LDS
        if (kk + 32 < H_) {
            ASYNC_CP16(gH0 + kk + 32, &hs[buf ^ 1][r0][s0 * 8]);
            ASYNC_CP16(gH1 + kk + 32, &hs[buf ^ 1][r1][s1 * 8]);
        }
#else
        uint4 a0 = *(const uint4*)(gH0 + kk);
        uint4 a1 = *(const uint4*)(gH1 + kk);
        __syncthreads();
        *(uint4*)&hs[0][r0][s0 * 8] = a0;
        *(uint4*)&hs[0][r1][s1 * 8] = a1;
        __syncthreads();
#endif

        BFrag fb;
        const __bf16* bp = wrow + kk + kh * 8;
        fb.q[0] = *(const uint4*)bp;
        fb.q[1] = *(const uint4*)(bp + 16);

        const int sb = USE_ASYNC_LDS ? buf : 0;
        #pragma unroll
        for (int mf = 0; mf < 4; ++mf) {
            BFrag fa;
            const __bf16* ap = &hs[sb][mf * 16 + row16][kh * 8];
            fa.q[0] = *(const uint4*)ap;
            fa.q[1] = *(const uint4*)(ap + 16);
            acc[mf].v = __builtin_amdgcn_wmma_f32_16x16x32_bf16(
                false, fa.v, false, fb.v, (short)0, acc[mf].v, false, false);
        }

#if USE_ASYNC_LDS
        ASYNC_WAIT();
        __syncthreads();
#endif
    }

    // park gate partials (h @ Wm^T) in LDS for the elementwise stage
    {
        const int nloc = lane & 15;
        const int mhi  = lane >> 4;
        #pragma unroll
        for (int mf = 0; mf < 4; ++mf)
            #pragma unroll
            for (int r = 0; r < 8; ++r)
                gbuf[wid][mf * 16 + mhi * 8 + r][nloc] = acc[mf].f[r];
    }
    __syncthreads();

    // fused cell update: 64 rows x 16 cols = 1024 items, 8 per thread
    for (int p = tid; p < 64 * 16; p += 128) {
        const int b  = p >> 4;
        const int jl = p & 15;
        const int j  = jbase + jl;
        const float* xp = XP + ((size_t)(t * B_ + b) << 12);
        const float gf = gbuf[0][b][jl] + xp[0 * H_ + j] + bm4[0 * H_ + j];
        const float gi = gbuf[1][b][jl] + xp[1 * H_ + j] + bm4[1 * H_ + j];
        const float go = gbuf[2][b][jl] + xp[2 * H_ + j] + bm4[2 * H_ + j];
        const float gc = gbuf[3][b][jl] + xp[3 * H_ + j] + bm4[3 * H_ + j];
        const float f  = 1.0f / (1.0f + __expf(-gf));
        const float i  = 1.0f / (1.0f + __expf(-gi));
        const float o  = 1.0f / (1.0f + __expf(-go));
        const float ct = tanhf(gc);
        const size_t sidx = (size_t)b * H_ + j;
        const float cn = f * cF[sidx] + i * ct;      // cell always updated
        const float hx = o * tanhf(cn);
        const float m  = (float)maskBL[b * L_ + t];  // mask on hidden only
        const float hn = hx * m + hF[sidx] * (1.0f - m);
        cF[sidx]   = cn;
        hF[sidx]   = hn;
        hBout[sidx] = (__bf16)hn;
        out[((size_t)b * L_ + t) * H_ + j] = hn;
    }
}

__global__ void finalize_kernel(const float* __restrict__ hF,
                                const float* __restrict__ cF,
                                float* __restrict__ tail, int n) {
    for (int i = blockIdx.x * blockDim.x + threadIdx.x; i < n;
         i += gridDim.x * blockDim.x) {
        tail[i]     = hF[i];
        tail[n + i] = cF[i];
    }
}

// ---------------------------------------------------------------------------
// Host-side orchestration (workspace ~622 MB; see layout below)
// ---------------------------------------------------------------------------
extern "C" void kernel_launch(void* const* d_in, const int* in_sizes, int n_in,
                              void* d_out, int out_size, void* d_ws, size_t ws_size,
                              hipStream_t stream) {
    const float* inputs  = (const float*)d_in[0];
    const int*   input_m = (const int*)  d_in[1];
    // setup_inputs order: per gate n in {f,i,o,c}: Wnx, bnx, Wnm, bnm
    const float* Wfx = (const float*)d_in[2];  const float* bfx = (const float*)d_in[3];
    const float* Wfm = (const float*)d_in[4];  const float* bfm = (const float*)d_in[5];
    const float* Wix = (const float*)d_in[6];  const float* bix = (const float*)d_in[7];
    const float* Wim = (const float*)d_in[8];  const float* bim = (const float*)d_in[9];
    const float* Wox = (const float*)d_in[10]; const float* box = (const float*)d_in[11];
    const float* Wom = (const float*)d_in[12]; const float* bom = (const float*)d_in[13];
    const float* Wcx = (const float*)d_in[14]; const float* bcx = (const float*)d_in[15];
    const float* Wcm = (const float*)d_in[16]; const float* bcm = (const float*)d_in[17];

    char* ws = (char*)d_ws;
    size_t off = 0;
    auto carve = [&](size_t bytes) {
        char* p = ws + off;
        off += (bytes + 255) & ~(size_t)255;
        return p;
    };
    __bf16* WXb = (__bf16*)carve((size_t)G_ * E_ * 2);        // 8 MB
    __bf16* WMb = (__bf16*)carve((size_t)G_ * H_ * 2);        // 8 MB
    __bf16* Xb  = (__bf16*)carve((size_t)B_ * L_ * E_ * 2);   // 64 MB
    float*  XP  = (float*) carve((size_t)L_ * B_ * G_ * 4);   // 512 MB
    float*  bx4 = (float*) carve((size_t)G_ * 4);
    float*  bm4 = (float*) carve((size_t)G_ * 4);
    float*  hF  = (float*) carve((size_t)B_ * H_ * 4);
    float*  cF  = (float*) carve((size_t)B_ * H_ * 4);
    __bf16* hB0 = (__bf16*)carve((size_t)B_ * H_ * 2);
    __bf16* hB1 = (__bf16*)carve((size_t)B_ * H_ * 2);

    float* out = (float*)d_out;                    // [B,L,H] ++ h_t ++ c_t
    float* tail = out + (size_t)B_ * L_ * H_;

    // 1) precision packing
    pack_x_kernel<<<2048, 256, 0, stream>>>(inputs, Xb, B_ * L_ * E_);
    pack_wb_kernel<<<2048, 256, 0, stream>>>(Wfx, Wix, Wox, Wcx,
                                             Wfm, Wim, Wom, Wcm,
                                             bfx, bix, box, bcx,
                                             bfm, bim, bom, bcm,
                                             WXb, WMb, bx4, bm4);
    init_state_kernel<<<256, 256, 0, stream>>>(hF, cF, hB0, hB1, B_ * H_);

    // 2) bulk input projection: [32768,1024] x [1024,4096] via WMMA bf16
    dim3 ggrid((B_ * L_) / 128, G_ / 128);
    gemm_xproj_kernel<<<ggrid, 256, 0, stream>>>(Xb, WXb, bx4, XP);

    // 3) 512 sequential recurrent steps (h ping-pongs between hB0/hB1)
    for (int t = 0; t < L_; ++t) {
        const __bf16* hin  = (t & 1) ? hB1 : hB0;
        __bf16*       hout = (t & 1) ? hB0 : hB1;
        lstm_step_kernel<<<H_ / 16, 128, 0, stream>>>(
            XP, WMb, bm4, input_m, hin, hout, hF, cF, out, t);
    }

    // 4) emit h_t, c_t
    finalize_kernel<<<256, 256, 0, stream>>>(hF, cF, tail, B_ * H_);
}
